// ATT_learner_22084721836662
// MI455X (gfx1250) — matrix operand order repaired and checked by reading, not compiled, not run on hardware
//
#include <hip/hip_runtime.h>
#include <hip/hip_bf16.h>
#include <stdint.h>

// ---------------------------------------------------------------------------
// Problem constants (reference: N=10240, D=256, K=30 -> keep top K+1=31/row)
// ---------------------------------------------------------------------------
#define NN 10240
#define DD 256

typedef _Float16 v16h __attribute__((ext_vector_type(16)));
typedef _Float16 v8h  __attribute__((ext_vector_type(8)));
typedef float    v8f  __attribute__((ext_vector_type(8)));

// ---------------------------------------------------------------------------
// Kernel 1: h = relu(x*w1)*w2 ; emb = h / max(||h||,eps), stored as f16.
// One 256-thread block per row (D==256, one element per thread).
// ---------------------------------------------------------------------------
__global__ void att_embed_kernel(const float* __restrict__ x,
                                 const float* __restrict__ w1,
                                 const float* __restrict__ w2,
                                 _Float16* __restrict__ emb) {
    __shared__ float red[256];
    const int row = blockIdx.x;
    const int tid = threadIdx.x;

    float v = x[(size_t)row * DD + tid];
    float h = fmaxf(v * w1[tid], 0.0f) * w2[tid];

    red[tid] = h * h;
    __syncthreads();
    #pragma unroll
    for (int s = 128; s > 0; s >>= 1) {
        if (tid < s) red[tid] += red[tid + s];
        __syncthreads();
    }
    const float inv = 1.0f / fmaxf(sqrtf(red[0]), 1e-12f);
    emb[(size_t)row * DD + tid] = (_Float16)(h * inv);
}

// ---------------------------------------------------------------------------
// Fragment loaders (wave32 layouts, ISA 7.12.2).
//   A (16x32 f16): lane L (hs=L>>4, r=L&15) = row M=r,
//     halves {kb+8*hs .. +7} and {kb+16+8*hs .. +7}      (2x b128)
//   B (32x16 f16): B[k,n] = emb[colBase+n, k] -> lane = emb row (colBase+r),
//     halves {kb+8*hs .. +7} and {kb+16+8*hs .. +7}      (2x b128)
// ---------------------------------------------------------------------------
__device__ __forceinline__ v16h load_a_frag(const _Float16* __restrict__ row,
                                            int kb, int hs) {
    v8h lo = *(const v8h*)(row + kb + hs * 8);
    v8h hi = *(const v8h*)(row + kb + 16 + hs * 8);
    v16h a;
    #pragma unroll
    for (int i = 0; i < 8; ++i) { a[i] = lo[i]; a[8 + i] = hi[i]; }
    return a;
}

__device__ __forceinline__ v16h load_b_frag(const _Float16* __restrict__ row,
                                            int kb, int hs) {
    v8h lo = *(const v8h*)(row + kb + hs * 8);
    v8h hi = *(const v8h*)(row + kb + 16 + hs * 8);
    v16h b;
    #pragma unroll
    for (int i = 0; i < 8; ++i) { b[i] = lo[i]; b[8 + i] = hi[i]; }
    return b;
}

// ---------------------------------------------------------------------------
// Kernel 2: sim = emb @ emb^T via v_wmma_f32_16x16x32_f16.
//
// Wave tile: 32 (M) x 64 (N) -> 8 f32 accumulators (2 M-subtiles x 4 N-subtiles).
// Per K-step: 12 b128 fragment loads feed 8 WMMAs (1.5 loads/WMMA).
// Fragments are double-buffered: K-step ks+1 loads are issued before the
// ks WMMAs, hiding L2 latency inside the matrix-pipe work.
// Block: 8 waves stacked in M -> 256 x 64 per block (B rows shared via WGP$).
// Grid: (NN/64, NN/256) = (160, 40). Exact tiling -> EXEC all-ones (WMMA req).
// ---------------------------------------------------------------------------
__global__ void att_sim_gemm_kernel(const _Float16* __restrict__ emb,
                                    float* __restrict__ out) {
    const int lane  = threadIdx.x & 31;
    const int wave  = threadIdx.x >> 5;
    const int r     = lane & 15;
    const int hs    = lane >> 4;
    const int mBase = (blockIdx.y * 8 + wave) * 32;   // 0..10208
    const int nBase = blockIdx.x * 64;                // 0..10176

    const _Float16* aRow0 = emb + (size_t)(mBase + r) * DD;
    const _Float16* aRow1 = emb + (size_t)(mBase + 16 + r) * DD;
    const _Float16* bRow0 = emb + (size_t)(nBase + r) * DD;
    const _Float16* bRow1 = emb + (size_t)(nBase + 16 + r) * DD;
    const _Float16* bRow2 = emb + (size_t)(nBase + 32 + r) * DD;
    const _Float16* bRow3 = emb + (size_t)(nBase + 48 + r) * DD;

    v8f acc[2][4];
    #pragma unroll
    for (int mi = 0; mi < 2; ++mi)
        #pragma unroll
        for (int j = 0; j < 4; ++j) acc[mi][j] = v8f{};

    v16h aF[2][2], bF[2][4];

    // Preload K-step 0
    aF[0][0] = load_a_frag(aRow0, 0, hs);
    aF[0][1] = load_a_frag(aRow1, 0, hs);
    bF[0][0] = load_b_frag(bRow0, 0, hs);
    bF[0][1] = load_b_frag(bRow1, 0, hs);
    bF[0][2] = load_b_frag(bRow2, 0, hs);
    bF[0][3] = load_b_frag(bRow3, 0, hs);

    #pragma unroll
    for (int ks = 0; ks < 8; ++ks) {
        const int cur = ks & 1;
        const int nxt = cur ^ 1;

        if (ks < 7) {               // issue next-step loads before compute
            const int kb = (ks + 1) * 32;
            aF[nxt][0] = load_a_frag(aRow0, kb, hs);
            aF[nxt][1] = load_a_frag(aRow1, kb, hs);
            bF[nxt][0] = load_b_frag(bRow0, kb, hs);
            bF[nxt][1] = load_b_frag(bRow1, kb, hs);
            bF[nxt][2] = load_b_frag(bRow2, kb, hs);
            bF[nxt][3] = load_b_frag(bRow3, kb, hs);
        }

        #pragma unroll
        for (int mi = 0; mi < 2; ++mi)
            #pragma unroll
            for (int j = 0; j < 4; ++j)
                acc[mi][j] = __builtin_amdgcn_wmma_f32_16x16x32_f16(
                    /*neg_a=*/false, aF[cur][mi], /*neg_b=*/false, bF[cur][j],
                    /*c_mod=*/(short)0, acc[mi][j],
                    /*reuse_a=*/false, /*reuse_b=*/false);
    }

    // C/D layout: VGPR v, lane -> (M = v + 8*hs, N = r) within each 16x16 tile
    #pragma unroll
    for (int mi = 0; mi < 2; ++mi) {
        const size_t rowBase = (size_t)(mBase + mi * 16 + 8 * hs);
        #pragma unroll
        for (int j = 0; j < 4; ++j) {
            const size_t col = (size_t)(nBase + j * 16 + r);
            #pragma unroll
            for (int v = 0; v < 8; ++v)
                out[(rowBase + v) * (size_t)NN + col] = acc[mi][j][v];
        }
    }
}

// ---------------------------------------------------------------------------
// Kernel 3: per-row top-(k+1) mask + relu, in place.
// One block (256 threads) per row. Row staged in LDS as order-preserving
// uint keys (40 KB). 32-step MSB-first radix select finds the (k+1)-th
// largest key (max t with count(key>=t) >= k+1; monotone predicate ->
// greedy bit build). Keep value iff key >= threshold and value > 0 (relu).
// ---------------------------------------------------------------------------
__device__ __forceinline__ uint32_t f2key(float f) {
    uint32_t u = __float_as_uint(f);
    return (u & 0x80000000u) ? ~u : (u | 0x80000000u);
}
__device__ __forceinline__ float key2f(uint32_t k) {
    uint32_t u = (k & 0x80000000u) ? (k & 0x7FFFFFFFu) : ~k;
    return __uint_as_float(u);
}

__global__ void att_topk_mask_kernel(float* __restrict__ sim,
                                     const int* __restrict__ kptr) {
    __shared__ uint32_t keys[NN];
    __shared__ uint32_t cnt;

    const int row = blockIdx.x;
    const int tid = threadIdx.x;
    float* rp = sim + (size_t)row * NN;

    for (int i = tid; i < NN; i += 256) keys[i] = f2key(rp[i]);

    const uint32_t kk = (uint32_t)(*kptr + 1);   // top (k+1) kept

    uint32_t prefix = 0;
    for (int b = 31; b >= 0; --b) {
        const uint32_t trial = prefix | (1u << b);
        if (tid == 0) cnt = 0;
        __syncthreads();                          // also covers keys[] fill
        uint32_t local = 0;
        for (int i = tid; i < NN; i += 256) local += (keys[i] >= trial) ? 1u : 0u;
        atomicAdd(&cnt, local);
        __syncthreads();
        if (cnt >= kk) prefix = trial;            // uniform across block
        __syncthreads();                          // protect cnt reset
    }

    for (int i = tid; i < NN; i += 256) {
        const uint32_t k = keys[i];
        const float f = key2f(k);
        rp[i] = (k >= prefix && f > 0.0f) ? f : 0.0f;
    }
}

// ---------------------------------------------------------------------------
// Launch
// ---------------------------------------------------------------------------
extern "C" void kernel_launch(void* const* d_in, const int* in_sizes, int n_in,
                              void* d_out, int out_size, void* d_ws, size_t ws_size,
                              hipStream_t stream) {
    const float* feat = (const float*)d_in[0];   // [N, D] f32
    const float* w1   = (const float*)d_in[1];   // [D]    f32
    const float* w2   = (const float*)d_in[2];   // [D]    f32
    const int*   kptr = (const int*)d_in[3];     // scalar k

    float*     out = (float*)d_out;              // [N, N] f32
    _Float16*  emb = (_Float16*)d_ws;            // [N, D] f16 (5.24 MB scratch)

    // 1) attentive MLP + L2 normalize -> f16 embeddings
    att_embed_kernel<<<NN, 256, 0, stream>>>(feat, w1, w2, emb);

    // 2) sim = emb @ emb^T via WMMA f16->f32 (32x64 per wave, double-buffered)
    dim3 g2(NN / 64, NN / 256);                  // (160, 40), 8 waves/block
    att_sim_gemm_kernel<<<g2, 256, 0, stream>>>(emb, out);

    // 3) per-row top-(k+1) mask + relu, in place
    att_topk_mask_kernel<<<NN, 256, 0, stream>>>(out, kptr);
}
